// FocalNet_2D_24300924961159
// MI455X (gfx1250) — compile-verified, matrix operand order checked by reading, not run on hardware
//
#include <hip/hip_runtime.h>
#include <hip/hip_bf16.h>

// ---------------- problem constants ----------------
constexpr int BB   = 8;
constexpr int HH   = 128;
constexpr int WW   = 128;
constexpr int CD   = 192;                 // channels
constexpr int FL   = 3;
constexpr int NF   = 2 * CD + FL + 1;     // 388
constexpr int LDF  = 400;                 // padded f stride (25 n-tiles of 16)
constexpr int HID  = 4 * CD;              // 768
constexpr long LTOT = (long)BB * HH * WW; // 131072 tokens

typedef __attribute__((ext_vector_type(16))) __bf16 v16bf;
typedef __attribute__((ext_vector_type(8)))  __bf16 v8bf;
typedef __attribute__((ext_vector_type(8)))  float  v8f;

// ---------------- workspace layout (bytes) ----------------
constexpr size_t OFF_FBUF  = 0;                                   // LTOT*400 f32
constexpr size_t OFF_CTXA  = OFF_FBUF + (size_t)LTOT * LDF * 4;   // LTOT*192 f32
constexpr size_t OFF_CTXB  = OFF_CTXA + (size_t)LTOT * CD * 4;
constexpr size_t OFF_CALL  = OFF_CTXB + (size_t)LTOT * CD * 4;    // ctx_all
constexpr size_t OFF_X1    = OFF_CALL + (size_t)LTOT * CD * 4;
constexpr size_t OFF_CG    = OFF_X1   + (size_t)LTOT * CD * 4;    // 8*192 f32
constexpr size_t OFF_FWPK  = OFF_CG   + 8192;                     // packed bf16 weights
constexpr size_t SZ_FWPK   = (size_t)25 * 6  * 512 * 2;
constexpr size_t OFF_HPK   = OFF_FWPK + SZ_FWPK;
constexpr size_t SZ_HPK    = (size_t)12 * 6  * 512 * 2;
constexpr size_t OFF_PPK   = OFF_HPK  + SZ_HPK;
constexpr size_t SZ_PPK    = (size_t)12 * 6  * 512 * 2;
constexpr size_t OFF_FC1PK = OFF_PPK  + SZ_PPK;
constexpr size_t SZ_FC1PK  = (size_t)48 * 6  * 512 * 2;
constexpr size_t OFF_FC2PK = OFF_FC1PK + SZ_FC1PK;

// ---------------- helpers ----------------
__device__ __forceinline__ float gelu_f(float x) {
    return 0.5f * x * (1.0f + erff(x * 0.70710678118654752f));
}
__device__ __forceinline__ float wave_sum(float s) {
    #pragma unroll
    for (int m = 16; m >= 1; m >>= 1) s += __shfl_xor(s, m, 32);
    return s;
}
__device__ __forceinline__ v8f wmma_bf16(v16bf a, v16bf b, v8f c) {
    return __builtin_amdgcn_wmma_f32_16x16x32_bf16(false, a, false, b, (short)0, c, false, false);
}
// A fragment from LDS, row-major bf16 [rows][lda]; 16-bit A layout:
// lane<16: M=lane, K = 0..7 & 16..23 ; lane>=16: M=lane-16, K = 8..15 & 24..31
__device__ __forceinline__ v16bf load_afrag(const __bf16* As, int lda, int m0, int kt, int lane) {
    int row = m0 + (lane & 15);
    int c0  = kt * 32 + ((lane >> 4) << 3);
    union { v16bf v; v8bf h[2]; } u;
    u.h[0] = *(const v8bf*)(As + row * lda + c0);
    u.h[1] = *(const v8bf*)(As + row * lda + c0 + 16);
    return u.v;
}
// B fragment from packed fragment-major weights
__device__ __forceinline__ v16bf load_bfrag(const __bf16* pk, int nt, int kt, int ktiles, int lane) {
    return *(const v16bf*)(pk + (((size_t)(nt * ktiles + kt)) * 32 + lane) * 16);
}

// ---------------- K0: pack weights into bf16 WMMA fragments ----------------
// B[k][n] = src[k*sk + n*sn], zero-padded beyond (K,N).
__global__ void pack_b_kernel(const float* __restrict__ src, __bf16* __restrict__ dst,
                              int K, int N, int sk, int sn, int ktiles) {
    int frag = blockIdx.x;
    int nt = frag / ktiles, kt = frag % ktiles;
    int lane = threadIdx.x;
    int n  = nt * 16 + (lane & 15);
    int hi = lane >> 4;
    #pragma unroll
    for (int j = 0; j < 16; ++j) {
        int klocal = (j < 8 ? j : j + 8) + hi * 8;
        int k = kt * 32 + klocal;
        float v = (k < K && n < N) ? src[(size_t)k * sk + (size_t)n * sn] : 0.0f;
        dst[((size_t)frag * 32 + lane) * 16 + j] = (__bf16)v;
    }
}

// ---------------- K1: LN1 + GEMM -> f (q | ctx | gates), stride 400 ----------------
__global__ __launch_bounds__(256) void ln1_gemm_f(
    const float* __restrict__ x, const float* __restrict__ g, const float* __restrict__ b,
    const __bf16* __restrict__ fw_pk, const float* __restrict__ f_b, float* __restrict__ fbuf) {
    __shared__ __attribute__((aligned(16))) __bf16 As[128 * CD];
    long t0 = (long)blockIdx.x * 128;
    int w = threadIdx.x >> 5, lane = threadIdx.x & 31;
    // LayerNorm: each wave normalizes 16 tokens
    for (int i = 0; i < 16; ++i) {
        int tl = w * 16 + i;
        const float* xr = x + (t0 + tl) * CD;
        float v[6]; float s = 0.f;
        #pragma unroll
        for (int j = 0; j < 6; ++j) { v[j] = xr[lane + 32 * j]; s += v[j]; }
        float mean = wave_sum(s) * (1.0f / CD);
        float q = 0.f;
        #pragma unroll
        for (int j = 0; j < 6; ++j) { float d = v[j] - mean; q += d * d; }
        float rs = rsqrtf(wave_sum(q) * (1.0f / CD) + 1e-5f);
        #pragma unroll
        for (int j = 0; j < 6; ++j) {
            int c = lane + 32 * j;
            As[tl * CD + c] = (__bf16)((v[j] - mean) * rs * g[c] + b[c]);
        }
    }
    __syncthreads();
    // GEMM: wave owns 16 rows, sweeps 25 n-tiles, K=192 (6 steps)
    int m0 = w * 16;
    v16bf a[6];
    #pragma unroll
    for (int kt = 0; kt < 6; ++kt) a[kt] = load_afrag(As, CD, m0, kt, lane);
    int col = lane & 15, rofs = (lane >> 4) * 8;
    for (int nt = 0; nt < 25; ++nt) {
        int n0 = nt * 16;
        float bv = (n0 + col < NF) ? f_b[n0 + col] : 0.0f;
        v8f acc;
        #pragma unroll
        for (int r = 0; r < 8; ++r) acc[r] = bv;
        #pragma unroll
        for (int kt = 0; kt < 6; ++kt)
            acc = wmma_bf16(a[kt], load_bfrag(fw_pk, nt, kt, 6, lane), acc);
        #pragma unroll
        for (int r = 0; r < 8; ++r)
            fbuf[(t0 + m0 + r + rofs) * LDF + n0 + col] = acc[r];
    }
}

// ---------------- K2: depthwise conv + GELU + gated accumulation ----------------
__global__ __launch_bounds__(256) void dwconv_gelu(
    const float* __restrict__ src, int sstride, int soff,
    const float* __restrict__ kern, int ks, int l,
    float* __restrict__ dst, float* __restrict__ ctx_all,
    const float* __restrict__ fbuf, int first) {
    long idx = (long)blockIdx.x * 256 + threadIdx.x;       // over LTOT*CD
    int c = (int)(idx % CD);
    long pix = idx / CD;
    int wq = (int)(pix % WW);
    int h  = (int)((pix / WW) % HH);
    long img0 = pix - ((long)h * WW + wq);                 // (b,0,0) pixel index
    int r = ks >> 1;
    float s = 0.f;
    for (int dy = -r; dy <= r; ++dy) {
        int hh = h + dy; if (hh < 0 || hh >= HH) continue;
        for (int dx = -r; dx <= r; ++dx) {
            int ww2 = wq + dx; if (ww2 < 0 || ww2 >= WW) continue;
            s += src[(img0 + (long)hh * WW + ww2) * sstride + soff + c] *
                 kern[(c * ks + (dy + r)) * ks + (dx + r)];
        }
    }
    float gl = gelu_f(s);
    dst[pix * CD + c] = gl;
    float upd = gl * fbuf[pix * LDF + 2 * CD + l];
    if (first) ctx_all[pix * CD + c] = upd;
    else       ctx_all[pix * CD + c] += upd;
}

// ---------------- K3: global context: gelu(mean over H,W) per (b,c) ----------------
__global__ __launch_bounds__(256) void ctx_mean(const float* __restrict__ ctx, float* __restrict__ cg) {
    int c = blockIdx.x % CD, b = blockIdx.x / CD;
    const float* p = ctx + ((long)b * HH * WW) * CD + c;
    float s = 0.f;
    for (int i = threadIdx.x; i < HH * WW; i += 256) s += p[(long)i * CD];
    __shared__ float red[256];
    red[threadIdx.x] = s; __syncthreads();
    for (int st = 128; st >= 1; st >>= 1) {
        if ((int)threadIdx.x < st) red[threadIdx.x] += red[threadIdx.x + st];
        __syncthreads();
    }
    if (threadIdx.x == 0) cg[blockIdx.x] = gelu_f(red[0] * (1.0f / (HH * WW)));
}

// ---------------- K4: modulator GEMM -> q*mod -> proj GEMM + shortcut -> x1 -------
__global__ __launch_bounds__(256) void mod_proj(
    const float* __restrict__ ctx_all, const float* __restrict__ cg,
    const float* __restrict__ fbuf,
    const __bf16* __restrict__ h_pk, const float* __restrict__ h_b,
    const __bf16* __restrict__ p_pk, const float* __restrict__ p_b,
    const float* __restrict__ x, float* __restrict__ x1) {
    __shared__ __attribute__((aligned(16))) __bf16 As[64 * CD];
    __shared__ __attribute__((aligned(16))) __bf16 Ms[64 * CD];
    long t0 = (long)blockIdx.x * 64;
    // stage ctx_all + ctx_global*gate3 as bf16
    for (int i = threadIdx.x; i < 64 * CD; i += 256) {
        int tl = i / CD, c = i % CD;
        long token = t0 + tl;
        int b = (int)(token >> 14);                        // /(H*W)
        float a = ctx_all[token * CD + c] + cg[b * CD + c] * fbuf[token * LDF + 2 * CD + FL];
        As[i] = (__bf16)a;
    }
    __syncthreads();
    int w = threadIdx.x >> 5, lane = threadIdx.x & 31;
    int m0 = (w & 3) * 16, nb = (w >> 2) * 6;
    int col = lane & 15, rofs = (lane >> 4) * 8;
    // GEMM1: modulator = As @ h_w^T + h_b, then Ms = bf16(q * mod)
    {
        v16bf a[6];
        #pragma unroll
        for (int kt = 0; kt < 6; ++kt) a[kt] = load_afrag(As, CD, m0, kt, lane);
        for (int nt2 = 0; nt2 < 6; ++nt2) {
            int nt = nb + nt2, n0 = nt * 16;
            float bv = h_b[n0 + col];
            v8f acc;
            #pragma unroll
            for (int r = 0; r < 8; ++r) acc[r] = bv;
            #pragma unroll
            for (int kt = 0; kt < 6; ++kt)
                acc = wmma_bf16(a[kt], load_bfrag(h_pk, nt, kt, 6, lane), acc);
            #pragma unroll
            for (int r = 0; r < 8; ++r) {
                int row = m0 + r + rofs;
                long token = t0 + row;
                float qv = fbuf[token * LDF + n0 + col];
                Ms[row * CD + n0 + col] = (__bf16)(qv * acc[r]);
            }
        }
    }
    __syncthreads();
    // GEMM2: x1 = Ms @ proj_w + proj_b + x
    {
        v16bf a[6];
        #pragma unroll
        for (int kt = 0; kt < 6; ++kt) a[kt] = load_afrag(Ms, CD, m0, kt, lane);
        for (int nt2 = 0; nt2 < 6; ++nt2) {
            int nt = nb + nt2, n0 = nt * 16;
            float bv = p_b[n0 + col];
            v8f acc;
            #pragma unroll
            for (int r = 0; r < 8; ++r) acc[r] = bv;
            #pragma unroll
            for (int kt = 0; kt < 6; ++kt)
                acc = wmma_bf16(a[kt], load_bfrag(p_pk, nt, kt, 6, lane), acc);
            #pragma unroll
            for (int r = 0; r < 8; ++r) {
                long o = (t0 + m0 + r + rofs) * CD + n0 + col;
                x1[o] = acc[r] + x[o];
            }
        }
    }
}

// ---------------- K5: LN2 + fc1 GEMM + GELU + fc2 GEMM + residual ----------------
__global__ __launch_bounds__(256) void mlp_fused(
    const float* __restrict__ x1, const float* __restrict__ g, const float* __restrict__ b,
    const __bf16* __restrict__ fc1_pk, const float* __restrict__ b1,
    const __bf16* __restrict__ fc2_pk, const float* __restrict__ b2,
    float* __restrict__ out) {
    __shared__ __attribute__((aligned(16))) __bf16 As[32 * CD];    // 12 KB
    __shared__ __attribute__((aligned(16))) __bf16 Hs[32 * HID];   // 48 KB
    long t0 = (long)blockIdx.x * 32;
    int w = threadIdx.x >> 5, lane = threadIdx.x & 31;
    // LN2: each wave handles 4 tokens
    for (int i = 0; i < 4; ++i) {
        int tl = w * 4 + i;
        const float* xr = x1 + (t0 + tl) * CD;
        float v[6]; float s = 0.f;
        #pragma unroll
        for (int j = 0; j < 6; ++j) { v[j] = xr[lane + 32 * j]; s += v[j]; }
        float mean = wave_sum(s) * (1.0f / CD);
        float q = 0.f;
        #pragma unroll
        for (int j = 0; j < 6; ++j) { float d = v[j] - mean; q += d * d; }
        float rs = rsqrtf(wave_sum(q) * (1.0f / CD) + 1e-5f);
        #pragma unroll
        for (int j = 0; j < 6; ++j) {
            int c = lane + 32 * j;
            As[tl * CD + c] = (__bf16)((v[j] - mean) * rs * g[c] + b[c]);
        }
    }
    __syncthreads();
    int m0 = (w & 1) * 16;
    int col = lane & 15, rofs = (lane >> 4) * 8;
    // GEMM1: hidden = gelu(As @ fc1 + b1); 48 n-tiles split 12 per wave-group
    {
        v16bf a[6];
        #pragma unroll
        for (int kt = 0; kt < 6; ++kt) a[kt] = load_afrag(As, CD, m0, kt, lane);
        int nb = (w >> 1) * 12;
        for (int nt2 = 0; nt2 < 12; ++nt2) {
            int nt = nb + nt2, n0 = nt * 16;
            float bv = b1[n0 + col];
            v8f acc;
            #pragma unroll
            for (int r = 0; r < 8; ++r) acc[r] = bv;
            #pragma unroll
            for (int kt = 0; kt < 6; ++kt)
                acc = wmma_bf16(a[kt], load_bfrag(fc1_pk, nt, kt, 6, lane), acc);
            #pragma unroll
            for (int r = 0; r < 8; ++r)
                Hs[(m0 + r + rofs) * HID + n0 + col] = (__bf16)gelu_f(acc[r]);
        }
    }
    __syncthreads();
    // GEMM2: out = x1 + Hs @ fc2 + b2 ; K=768 (24 steps), 3 n-tiles per wave
    {
        int nb2 = (w >> 1) * 3;
        v8f acc[3];
        #pragma unroll
        for (int j = 0; j < 3; ++j) {
            float bv = b2[(nb2 + j) * 16 + col];
            #pragma unroll
            for (int r = 0; r < 8; ++r) acc[j][r] = bv;
        }
        for (int kt = 0; kt < 24; ++kt) {
            v16bf af = load_afrag(Hs, HID, m0, kt, lane);
            #pragma unroll
            for (int j = 0; j < 3; ++j)
                acc[j] = wmma_bf16(af, load_bfrag(fc2_pk, nb2 + j, kt, 24, lane), acc[j]);
        }
        #pragma unroll
        for (int j = 0; j < 3; ++j)
            #pragma unroll
            for (int r = 0; r < 8; ++r) {
                long o = (t0 + m0 + r + rofs) * CD + (nb2 + j) * 16 + col;
                out[o] = x1[o] + acc[j][r];
            }
    }
}

// ---------------- launcher ----------------
extern "C" void kernel_launch(void* const* d_in, const int* in_sizes, int n_in,
                              void* d_out, int out_size, void* d_ws, size_t ws_size,
                              hipStream_t stream) {
    const float* x      = (const float*)d_in[0];
    const float* ln1_g  = (const float*)d_in[1];
    const float* ln1_b  = (const float*)d_in[2];
    const float* f_w    = (const float*)d_in[3];
    const float* f_b    = (const float*)d_in[4];
    const float* fk0    = (const float*)d_in[5];
    const float* fk1    = (const float*)d_in[6];
    const float* fk2    = (const float*)d_in[7];
    const float* h_w    = (const float*)d_in[8];
    const float* h_b    = (const float*)d_in[9];
    const float* proj_w = (const float*)d_in[10];
    const float* proj_b = (const float*)d_in[11];
    const float* ln2_g  = (const float*)d_in[12];
    const float* ln2_b  = (const float*)d_in[13];
    const float* fc1_w  = (const float*)d_in[14];
    const float* fc1_b  = (const float*)d_in[15];
    const float* fc2_w  = (const float*)d_in[16];
    const float* fc2_b  = (const float*)d_in[17];
    float* out = (float*)d_out;

    char* ws = (char*)d_ws;
    float*  fbuf    = (float*)(ws + OFF_FBUF);
    float*  ctxA    = (float*)(ws + OFF_CTXA);
    float*  ctxB    = (float*)(ws + OFF_CTXB);
    float*  ctx_all = (float*)(ws + OFF_CALL);
    float*  x1      = (float*)(ws + OFF_X1);
    float*  cg      = (float*)(ws + OFF_CG);
    __bf16* fw_pk   = (__bf16*)(ws + OFF_FWPK);
    __bf16* h_pk    = (__bf16*)(ws + OFF_HPK);
    __bf16* p_pk    = (__bf16*)(ws + OFF_PPK);
    __bf16* fc1_pk  = (__bf16*)(ws + OFF_FC1PK);
    __bf16* fc2_pk  = (__bf16*)(ws + OFF_FC2PK);

    // K0: pack weights (fragment-major bf16)
    pack_b_kernel<<<25 * 6, 32, 0, stream>>>(f_w,    fw_pk,  CD,  NF, NF,  1,   6);
    pack_b_kernel<<<12 * 6, 32, 0, stream>>>(h_w,    h_pk,   CD,  CD, 1,   CD,  6);   // B = h_w^T
    pack_b_kernel<<<12 * 6, 32, 0, stream>>>(proj_w, p_pk,   CD,  CD, CD,  1,   6);
    pack_b_kernel<<<48 * 6, 32, 0, stream>>>(fc1_w,  fc1_pk, CD,  HID, HID, 1,  6);
    pack_b_kernel<<<12 * 24, 32, 0, stream>>>(fc2_w, fc2_pk, HID, CD, CD,  1,   24);

    // K1: LN1 + f GEMM
    ln1_gemm_f<<<(int)(LTOT / 128), 256, 0, stream>>>(x, ln1_g, ln1_b, fw_pk, f_b, fbuf);

    // K2: focal levels
    int cgrid = (int)((LTOT * CD) / 256);
    dwconv_gelu<<<cgrid, 256, 0, stream>>>(fbuf, LDF, CD, fk0, 3, 0, ctxA, ctx_all, fbuf, 1);
    dwconv_gelu<<<cgrid, 256, 0, stream>>>(ctxA, CD, 0,  fk1, 5, 1, ctxB, ctx_all, fbuf, 0);
    dwconv_gelu<<<cgrid, 256, 0, stream>>>(ctxB, CD, 0,  fk2, 7, 2, ctxA, ctx_all, fbuf, 0);

    // K3: global context (from final ctx = ctxA)
    ctx_mean<<<BB * CD, 256, 0, stream>>>(ctxA, cg);

    // K4: modulator + q*mod + proj + shortcut
    mod_proj<<<(int)(LTOT / 64), 256, 0, stream>>>(ctx_all, cg, fbuf, h_pk, h_b, p_pk, proj_b, x, x1);

    // K5: LN2 + MLP + residual
    mlp_fused<<<(int)(LTOT / 32), 256, 0, stream>>>(x1, ln2_g, ln2_b, fc1_pk, fc1_b, fc2_pk, fc2_b, out);
}
